// MultiHeadAttention_11441792876738
// MI455X (gfx1250) — matrix operand order repaired and checked by reading, not compiled, not run on hardware
//
#include <hip/hip_runtime.h>

// Problem constants (match reference)
#define HN   16          // heads
#define HS   64          // head size
#define CEMB 1024        // embedding dim
#define BB   4
#define TT   2048
#define BT   (BB * TT)   // 8192 rows

typedef __bf16 bf16;
typedef __attribute__((ext_vector_type(8)))  __bf16 v8bf;
typedef __attribute__((ext_vector_type(16))) __bf16 v16bf;
typedef __attribute__((ext_vector_type(8)))  float  v8f;

static __device__ inline v8f wmma_bf16(v16bf a, v16bf b, v8f c) {
  // v_wmma_f32_16x16x32_bf16  D = A(16x32) * B(32x16) + C
  return __builtin_amdgcn_wmma_f32_16x16x32_bf16(false, a, false, b,
                                                 (short)0, c, false, false);
}

// A-matrix fragment, 16x32 bf16, ISA layout:
// lane l<16 : row l,    K = 0..7 (v0-3) and 16..23 (v4-7)
// lane l>=16: row l-16, K = 8..15          and 24..31
static __device__ inline v16bf load_fragA(const bf16* base, int stride) {
  const int lane = threadIdx.x & 31;
  const int r = lane & 15, sel = lane >> 4;
  const bf16* p = base + (size_t)r * stride + sel * 8;
  v8bf lo = *(const v8bf*)(p);
  v8bf hi = *(const v8bf*)(p + 16);
  return __builtin_shufflevector(lo, hi, 0, 1, 2, 3, 4, 5, 6, 7,
                                 8, 9, 10, 11, 12, 13, 14, 15);
}

// B-matrix fragment, 32x16 bf16, from Bt stored (N,K) row-major:
// lane l: column n = l&15, K = (l>>4)*8 .. +7 and 16+(l>>4)*8 .. +7
static __device__ inline v16bf load_fragB(const bf16* base, int stride) {
  const int lane = threadIdx.x & 31;
  const bf16* p = base + (size_t)(lane & 15) * stride + (lane >> 4) * 8;
  v8bf lo = *(const v8bf*)(p);
  v8bf hi = *(const v8bf*)(p + 16);
  return __builtin_shufflevector(lo, hi, 0, 1, 2, 3, 4, 5, 6, 7,
                                 8, 9, 10, 11, 12, 13, 14, 15);
}

// ---------------- conversion kernels ----------------
__global__ __launch_bounds__(256) void k_f32_to_bf16(const float* __restrict__ s,
                                                     bf16* __restrict__ d, int n) {
  int i = blockIdx.x * 256 + threadIdx.x;
  if (i < n) d[i] = (bf16)s[i];
}

// Wt[n,k] = (bf16) W[k,n]   (1024x1024)
__global__ __launch_bounds__(256) void k_wtrans(const float* __restrict__ s,
                                                bf16* __restrict__ d) {
  int i = blockIdx.x * 256 + threadIdx.x;   // < 1M
  int n = i >> 10, k = i & 1023;
  d[i] = (bf16)s[k * 1024 + n];
}

// permutation: M[b].flat[F] = O[b].flat[(F%T)*C + F/T]
__global__ __launch_bounds__(256) void k_permute(const bf16* __restrict__ O,
                                                 bf16* __restrict__ M) {
  int i = blockIdx.x * 256 + threadIdx.x;   // < BT*CEMB = 2^23
  int b = i >> 21;                          // TT*CEMB = 2^21
  int F = i & ((1 << 21) - 1);
  int src = (b << 21) + ((F & (TT - 1)) << 10) + (F >> 11);
  M[i] = O[src];
}

// ---------------- WMMA GEMM: C(M,N) = A(M,K) * Bt(N,K)^T ----------------
// Wave tile 32(M) x 64(N): 8 WMMAs per 32-deep K-step, 12 b128 loads.
// MODE 0: store bf16 head-split (B,H,T,hs)     (K projection)
// MODE 1: same, scaled by 1/sqrt(hs)           (Q projection)
// MODE 2: store bf16 transposed (B,H,hs,T)     (V projection)
// MODE 3: store fp32 row-major (M,N)           (output projection)
template <int MODE>
static __device__ inline void store_tile(v8f c, int m0, int n0, int N,
                                         void* __restrict__ out) {
  const int lane = threadIdx.x & 31;
  const int sel = lane >> 4, cl = lane & 15;
  const int col = n0 + cl;
#pragma unroll
  for (int j = 0; j < 8; ++j) {
    const int row = m0 + j + 8 * sel;
    float v = c[j];
    if (MODE == 3) {
      ((float*)out)[(size_t)row * N + col] = v;
    } else {
      const int b = row >> 11, t = row & (TT - 1);
      const int h = col >> 6, d = col & (HS - 1);
      bf16* o = (bf16*)out;
      if (MODE == 2)
        o[(size_t)(b * HN + h) * (HS * TT) + d * TT + t] = (bf16)v;
      else
        o[(size_t)(b * HN + h) * (TT * HS) + t * HS + d] =
            (bf16)(MODE == 1 ? v * 0.125f : v);
    }
  }
}

template <int MODE>
__global__ __launch_bounds__(256) void k_gemm_bf16(const bf16* __restrict__ A,
                                                   const bf16* __restrict__ Bt,
                                                   void* __restrict__ Out,
                                                   int M, int N, int K) {
  const int wave = threadIdx.x >> 5;
  // 8 waves: 4 along M (32 each), 2 along N (64 each) -> block tile 128x128
  const int m0 = blockIdx.y * 128 + (wave >> 1) * 32;
  const int n0 = blockIdx.x * 128 + (wave & 1) * 64;

  const v8f vzero = {0.f, 0.f, 0.f, 0.f, 0.f, 0.f, 0.f, 0.f};
  v8f acc[2][4];
#pragma unroll
  for (int i = 0; i < 2; ++i)
#pragma unroll
    for (int j = 0; j < 4; ++j) acc[i][j] = vzero;

  const bf16* ap0 = A + (size_t)m0 * K;
  const bf16* ap1 = A + (size_t)(m0 + 16) * K;
  const bf16* bp0 = Bt + (size_t)n0 * K;
  const bf16* bp1 = Bt + (size_t)(n0 + 16) * K;
  const bf16* bp2 = Bt + (size_t)(n0 + 32) * K;
  const bf16* bp3 = Bt + (size_t)(n0 + 48) * K;

  for (int kk = 0; kk < K; kk += 32) {
    // L2-resident streaming; prefetch ahead (global_prefetch_b8)
    __builtin_prefetch(ap0 + kk + 256, 0, 0);
    __builtin_prefetch(bp0 + kk + 256, 0, 0);
    v16bf a0 = load_fragA(ap0 + kk, K);
    v16bf a1 = load_fragA(ap1 + kk, K);
    v16bf b0 = load_fragB(bp0 + kk, K);
    v16bf b1 = load_fragB(bp1 + kk, K);
    v16bf b2 = load_fragB(bp2 + kk, K);
    v16bf b3 = load_fragB(bp3 + kk, K);
    acc[0][0] = wmma_bf16(a0, b0, acc[0][0]);
    acc[0][1] = wmma_bf16(a0, b1, acc[0][1]);
    acc[0][2] = wmma_bf16(a0, b2, acc[0][2]);
    acc[0][3] = wmma_bf16(a0, b3, acc[0][3]);
    acc[1][0] = wmma_bf16(a1, b0, acc[1][0]);
    acc[1][1] = wmma_bf16(a1, b1, acc[1][1]);
    acc[1][2] = wmma_bf16(a1, b2, acc[1][2]);
    acc[1][3] = wmma_bf16(a1, b3, acc[1][3]);
  }
#pragma unroll
  for (int i = 0; i < 2; ++i)
#pragma unroll
    for (int j = 0; j < 4; ++j)
      store_tile<MODE>(acc[i][j], m0 + 16 * i, n0 + 16 * j, N, Out);
}

// ---------------- flash attention (causal, online softmax) ----------------
// One wave handles one 16-query tile of one (b,h). Q/K bf16 (B,H,T,hs),
// V bf16 transposed (B,H,hs,T). O bf16 (B,H,T,hs).
__global__ __launch_bounds__(256) void k_attn(const bf16* __restrict__ Q,
                                              const bf16* __restrict__ Kb,
                                              const bf16* __restrict__ Vt,
                                              bf16* __restrict__ O) {
  __shared__ v8bf plds_v[8][64];  // per-wave 16x32 bf16 P tile (1KB each)
  const int wave = threadIdx.x >> 5;
  const int lane = threadIdx.x & 31;
  const int sel = lane >> 4, cl = lane & 15;
  const int gid = blockIdx.x * 8 + wave;  // 0..8191
  const int bh = gid >> 7;                // (b*HN + h)
  const int qt = gid & 127;               // query tile (16 rows)

  const bf16* qbase = Q + ((size_t)bh * TT + qt * 16) * HS;
  const bf16* kbase = Kb + (size_t)bh * TT * HS;
  const bf16* vbase = Vt + (size_t)bh * HS * TT;
  bf16* pl = (bf16*)&plds_v[wave][0];

  const v16bf qf0 = load_fragA(qbase, HS);        // dims 0..31
  const v16bf qf1 = load_fragA(qbase + 32, HS);   // dims 32..63

  const v8f vzero = {0.f, 0.f, 0.f, 0.f, 0.f, 0.f, 0.f, 0.f};
  float mrow[8], lrow[8];
  v8f oacc[4];
#pragma unroll
  for (int j = 0; j < 8; ++j) { mrow[j] = -3.0e38f; lrow[j] = 0.f; }
#pragma unroll
  for (int d = 0; d < 4; ++d) oacc[d] = vzero;

  const int nch = (qt * 16 + 15) / 32 + 1;  // causal key-chunk count
  for (int ch = 0; ch < nch; ++ch) {
    const int k0 = ch * 32;
    v8f s0 = vzero, s1 = vzero;  // scores for keys k0..+15 / k0+16..+31
    s0 = wmma_bf16(qf0, load_fragB(kbase + (size_t)k0 * HS, HS), s0);
    s0 = wmma_bf16(qf1, load_fragB(kbase + (size_t)k0 * HS + 32, HS), s0);
    s1 = wmma_bf16(qf0, load_fragB(kbase + (size_t)(k0 + 16) * HS, HS), s1);
    s1 = wmma_bf16(qf1, load_fragB(kbase + (size_t)(k0 + 16) * HS + 32, HS), s1);

#pragma unroll
    for (int j = 0; j < 8; ++j) {
      const int row = qt * 16 + j + 8 * sel;
      float v0 = s0[j]; if (k0 + cl > row)      v0 = -3.0e38f;  // causal mask
      float v1 = s1[j]; if (k0 + 16 + cl > row) v1 = -3.0e38f;
      // row-max across the 16 lanes holding this row
      float mx = fmaxf(v0, v1);
      mx = fmaxf(mx, __shfl_xor(mx, 1, 16));
      mx = fmaxf(mx, __shfl_xor(mx, 2, 16));
      mx = fmaxf(mx, __shfl_xor(mx, 4, 16));
      mx = fmaxf(mx, __shfl_xor(mx, 8, 16));
      const float mnew = fmaxf(mrow[j], mx);
      const float alpha = __expf(mrow[j] - mnew);
      mrow[j] = mnew;
      const float p0 = __expf(v0 - mnew);
      const float p1 = __expf(v1 - mnew);
      float ps = p0 + p1;
      ps += __shfl_xor(ps, 1, 16);
      ps += __shfl_xor(ps, 2, 16);
      ps += __shfl_xor(ps, 4, 16);
      ps += __shfl_xor(ps, 8, 16);
      lrow[j] = lrow[j] * alpha + ps;
#pragma unroll
      for (int d = 0; d < 4; ++d) oacc[d][j] *= alpha;
      // stage P (C-layout -> row-major 16x32) in wave-private LDS
      pl[(j + 8 * sel) * 32 + cl] = (bf16)p0;
      pl[(j + 8 * sel) * 32 + cl + 16] = (bf16)p1;
    }
    // intra-wave LDS ordering (LDS is in-order per wave; fence the compiler
    // and drain DScnt before re-reading as an A-fragment)
    asm volatile("s_wait_dscnt 0" ::: "memory");
    const v16bf pf = load_fragA(pl, 32);
#pragma unroll
    for (int d = 0; d < 4; ++d) {
      v16bf vf = load_fragB(vbase + (size_t)(d * 16) * TT + k0, TT);
      oacc[d] = wmma_bf16(pf, vf, oacc[d]);
    }
  }

  bf16* obase = O + ((size_t)bh * TT + qt * 16) * HS;
  float rinv[8];
#pragma unroll
  for (int j = 0; j < 8; ++j) rinv[j] = 1.0f / lrow[j];
#pragma unroll
  for (int d = 0; d < 4; ++d)
#pragma unroll
    for (int j = 0; j < 8; ++j) {
      float v = oacc[d][j] * rinv[j];
      obase[(j + 8 * sel) * HS + d * 16 + cl] = (bf16)v;
    }
}

// ---------------- host-side launch ----------------
extern "C" void kernel_launch(void* const* d_in, const int* in_sizes, int n_in,
                              void* d_out, int out_size, void* d_ws, size_t ws_size,
                              hipStream_t stream) {
  const float* x  = (const float*)d_in[0];
  const float* Wq = (const float*)d_in[1];
  const float* Wk = (const float*)d_in[2];
  const float* Wv = (const float*)d_in[3];
  const float* Wp = (const float*)d_in[4];

  char* ws = (char*)d_ws;
  const size_t MB = 1024ull * 1024ull;
  bf16* wqt = (bf16*)(ws + 0 * MB);   // 2MB each, (N,K) bf16
  bf16* wkt = (bf16*)(ws + 2 * MB);
  bf16* wvt = (bf16*)(ws + 4 * MB);
  bf16* wpt = (bf16*)(ws + 6 * MB);
  bf16* xb  = (bf16*)(ws + 8 * MB);   // 16MB  (dead after QKV GEMMs)
  bf16* Ob  = (bf16*)(ws + 8 * MB);   // alias xb
  bf16* qb  = (bf16*)(ws + 24 * MB);  // 16MB  (dead after attention)
  bf16* Mp  = (bf16*)(ws + 24 * MB);  // alias qb
  bf16* kb  = (bf16*)(ws + 40 * MB);  // 16MB
  bf16* vtb = (bf16*)(ws + 56 * MB);  // 16MB   (total 72MB)

  const int nx = BT * CEMB;  // 8388608

  k_f32_to_bf16<<<nx / 256, 256, 0, stream>>>(x, xb, nx);
  k_wtrans<<<(CEMB * CEMB) / 256, 256, 0, stream>>>(Wq, wqt);
  k_wtrans<<<(CEMB * CEMB) / 256, 256, 0, stream>>>(Wk, wkt);
  k_wtrans<<<(CEMB * CEMB) / 256, 256, 0, stream>>>(Wv, wvt);
  k_wtrans<<<(CEMB * CEMB) / 256, 256, 0, stream>>>(Wp, wpt);

  dim3 gg(CEMB / 128, BT / 128);  // (8, 64)
  k_gemm_bf16<1><<<gg, 256, 0, stream>>>(xb, wqt, qb, BT, CEMB, CEMB);  // Q*0.125
  k_gemm_bf16<0><<<gg, 256, 0, stream>>>(xb, wkt, kb, BT, CEMB, CEMB);  // K
  k_gemm_bf16<2><<<gg, 256, 0, stream>>>(xb, wvt, vtb, BT, CEMB, CEMB); // V^T

  k_attn<<<1024, 256, 0, stream>>>(qb, kb, vtb, Ob);

  k_permute<<<nx / 256, 256, 0, stream>>>(Ob, Mp);
  k_gemm_bf16<3><<<gg, 256, 0, stream>>>(Mp, wpt, d_out, BT, CEMB, CEMB);
}
